// NystromAttention_738734375521
// MI455X (gfx1250) — matrix operand order repaired and checked by reading, not compiled, not run on hardware
//
#include <hip/hip_runtime.h>
#include <hip/hip_bf16.h>

typedef __attribute__((ext_vector_type(16))) _Float16 v16h;
typedef __attribute__((ext_vector_type(8)))  _Float16 v8h;
typedef __attribute__((ext_vector_type(8)))  float    v8f;

#define BH       64
#define SEQ      8192
#define DD       64
#define MM       128
#define NS_ITERS 12
#define NSPLIT   8

#define WMMA_F32_F16(a,b,c) \
  __builtin_amdgcn_wmma_f32_16x16x32_f16(false,(a),false,(b),(short)0,(c),false,false)

// ---- WMMA fragment loaders (CDNA5 16-bit layouts, wave32) -------------------

// A-matrix 16x32 tile from row-major [rows][ld] f16.
// lane<16: K = k0+0..15; lane>=16: K = k0+16..31.
__device__ inline v16h loadA16(const _Float16* base, int ld, int m0, int k0, int lane) {
  const _Float16* p = base + (m0 + (lane & 15)) * ld + k0 + ((lane & 16) ? 16 : 0);
  return *(const v16h*)p;
}

// B-matrix 32x16 tile where B = X^T, X row-major [cols][ld] f16 (contiguous).
// lane = output column j; half h -> K = k0 + 16*(lane>=16) + h.
__device__ inline v16h loadBt16(const _Float16* X, int ld, int j0, int k0, int lane) {
  const _Float16* p = X + (j0 + (lane & 15)) * ld + k0 + ((lane & 16) ? 16 : 0);
  return *(const v16h*)p;
}

// B-matrix 32x16 tile from non-transposed row-major X[k][n] (strided gather).
// MODE==1 applies (2I - X) on the fly (Newton-Schulz step).
template<int MODE>
__device__ inline v16h loadBn16(const _Float16* X, int ld, int k0, int n0, int lane) {
  int n  = n0 + (lane & 15);
  int kb = k0 + ((lane & 16) ? 16 : 0);
  v16h b;
#pragma unroll
  for (int h = 0; h < 16; ++h) {
    float x = (float)X[(kb + h) * ld + n];
    if (MODE == 1) x = (((kb + h) == n) ? 2.0f : 0.0f) - x;
    b[h] = (_Float16)x;
  }
  return b;
}

// 128x128x128 f16 GEMM within one 256-thread block (8 waves, 8 tiles/wave).
template<int MODE>
__device__ inline void mm128(_Float16* __restrict__ out, const _Float16* __restrict__ A,
                             const _Float16* __restrict__ X, int wave, int lane) {
  int m0 = wave * 16;
  int mo = (lane & 16) ? 8 : 0;
  v16h a[4];
#pragma unroll
  for (int kk = 0; kk < 4; ++kk) a[kk] = loadA16(A, 128, m0, kk * 32, lane);
  for (int tn = 0; tn < 8; ++tn) {
    v8f c = {};
#pragma unroll
    for (int kk = 0; kk < 4; ++kk)
      c = WMMA_F32_F16(a[kk], loadBn16<MODE>(X, 128, kk * 32, tn * 16, lane), c);
    int n = tn * 16 + (lane & 15);
#pragma unroll
    for (int e = 0; e < 8; ++e) out[(m0 + e + mo) * 128 + n] = (_Float16)c[e];
  }
}

// ---- kernels ---------------------------------------------------------------

__global__ void k_zero(float* p, int n) {
  int i = blockIdx.x * 256 + threadIdx.x;
  if (i < n) p[i] = 0.f;
}

// mean-pool 64 consecutive rows -> 128 landmarks; q is pre-scaled by 1/8.
__global__ void k_landmarks(const float* __restrict__ q, const float* __restrict__ k,
                            _Float16* __restrict__ qlg, _Float16* __restrict__ klg) {
  int b  = blockIdx.x;
  int bh = b >> 5;
  int m  = (b & 31) * 4 + (threadIdx.x >> 6);
  int d  = threadIdx.x & 63;
  const float* qp = q + ((size_t)bh * SEQ + m * 64) * DD + d;
  const float* kp = k + ((size_t)bh * SEQ + m * 64) * DD + d;
  float sq = 0.f, sk = 0.f;
#pragma unroll 8
  for (int j = 0; j < 64; ++j) { sq += qp[j * DD]; sk += kp[j * DD]; }
  qlg[bh * MM * DD + m * DD + d] = (_Float16)(sq * (0.125f / 64.f));
  klg[bh * MM * DD + m * DD + d] = (_Float16)(sk * (1.f / 64.f));
}

// BV = softmax(q_land @ k^T) @ v, streamed over n in tiles of 32 keys.
__global__ void k_bv(const float* __restrict__ k, const float* __restrict__ v,
                     const _Float16* __restrict__ qlg,
                     float* __restrict__ BVacc, float* __restrict__ rsg) {
  __shared__ _Float16 kt[32 * 64];   // k tile, row-major [n][d]
  __shared__ _Float16 vT[64 * 32];   // v tile transposed [d][n]
  __shared__ _Float16 P [128 * 32];  // exp(S) tile
  int tid = threadIdx.x, lane = tid & 31, wave = tid >> 5;
  int bh = blockIdx.x / NSPLIT, sp = blockIdx.x % NSPLIT;
  const float* kb = k + (size_t)bh * SEQ * DD;
  const float* vb = v + (size_t)bh * SEQ * DD;
  const _Float16* ql = qlg + bh * MM * DD;

  int m0 = wave * 16, mo = (lane & 16) ? 8 : 0;
  v16h aq0 = loadA16(ql, DD, m0, 0, lane);
  v16h aq1 = loadA16(ql, DD, m0, 32, lane);
  v8f acc[4] = {};
  float rsLocal = 0.f;

  int nbase = sp * (SEQ / NSPLIT);
  for (int itn = 0; itn < (SEQ / NSPLIT) / 32; ++itn) {
    int n0 = nbase + itn * 32;
    {   // stage k tile and transposed v tile (f32 -> f16)
      int nn = tid >> 3, d0 = (tid & 7) * 8;
#pragma unroll
      for (int i = 0; i < 8; ++i) {
        kt[nn * 64 + d0 + i]   = (_Float16)kb[(size_t)(n0 + nn) * DD + d0 + i];
        vT[(d0 + i) * 32 + nn] = (_Float16)vb[(size_t)(n0 + nn) * DD + d0 + i];
      }
    }
    __syncthreads();
    // S = q_land @ k_tile^T -> exp -> P  (128x32)
#pragma unroll
    for (int tn = 0; tn < 2; ++tn) {
      v8f c = {};
      c = WMMA_F32_F16(aq0, loadBt16(kt, 64, tn * 16, 0, lane), c);
      c = WMMA_F32_F16(aq1, loadBt16(kt, 64, tn * 16, 32, lane), c);
      int n = tn * 16 + (lane & 15);
#pragma unroll
      for (int e = 0; e < 8; ++e) P[(m0 + e + mo) * 32 + n] = (_Float16)__expf(c[e]);
    }
    __syncthreads();
    if (tid < 128) {          // row sums of this P tile
      float s = 0.f;
#pragma unroll
      for (int j = 0; j < 32; ++j) s += (float)P[tid * 32 + j];
      rsLocal += s;
    }
    // BV += P @ v_tile  (K = 32, single wmma step per output tile)
    v16h ap = loadA16(P, 32, m0, 0, lane);
#pragma unroll
    for (int tn = 0; tn < 4; ++tn)
      acc[tn] = WMMA_F32_F16(ap, loadBt16(vT, 32, tn * 16, 0, lane), acc[tn]);
    __syncthreads();
  }
  if (tid < 128) atomicAdd(&rsg[bh * MM + tid], rsLocal);
#pragma unroll
  for (int tn = 0; tn < 4; ++tn) {
    int n = tn * 16 + (lane & 15);
#pragma unroll
    for (int e = 0; e < 8; ++e)
      atomicAdd(&BVacc[bh * MM * DD + (m0 + e + mo) * DD + n], acc[tn][e]);
  }
}

__global__ void k_bvfin(const float* __restrict__ BVacc, const float* __restrict__ rsg,
                        _Float16* __restrict__ BVg) {
  int i  = blockIdx.x * 256 + threadIdx.x;   // < 64*128*64
  int bh = i >> 13;
  int m  = (i >> 6) & 127;
  BVg[i] = (_Float16)(BVacc[i] / rsg[bh * MM + m]);
}

// A = softmax(q_land @ k_land^T); A+ via Newton-Schulz (Z <- Z(2I - AZ));
// then G = A+ @ BV written transposed (Gt[d][m]) for contiguous B-frag loads.
__global__ void k_amat(const _Float16* __restrict__ qlg, const _Float16* __restrict__ klg,
                       const _Float16* __restrict__ BVg, _Float16* __restrict__ Gtg) {
  __shared__ _Float16 Kr  [128 * 128];
  __shared__ _Float16 BufA[128 * 128];
  __shared__ _Float16 BufB[128 * 128];
  __shared__ _Float16 BufC[128 * 128];
  __shared__ float rowsum[128];
  __shared__ float colsum[128];
  __shared__ float sscal;
  int tid = threadIdx.x, lane = tid & 31, wave = tid >> 5;
  int bh = blockIdx.x;
  const _Float16* ql = qlg + bh * MM * DD;
  const _Float16* kl = klg + bh * MM * DD;
  int m0 = wave * 16, mo = (lane & 16) ? 8 : 0;

  {   // exp(q_land @ k_land^T), operands straight from global (L2-resident)
    v16h a0 = loadA16(ql, DD, m0, 0, lane);
    v16h a1 = loadA16(ql, DD, m0, 32, lane);
    for (int tn = 0; tn < 8; ++tn) {
      v8f c = {};
      c = WMMA_F32_F16(a0, loadBt16(kl, DD, tn * 16, 0, lane), c);
      c = WMMA_F32_F16(a1, loadBt16(kl, DD, tn * 16, 32, lane), c);
      int n = tn * 16 + (lane & 15);
#pragma unroll
      for (int e = 0; e < 8; ++e) Kr[(m0 + e + mo) * 128 + n] = (_Float16)__expf(c[e]);
    }
  }
  __syncthreads();
  if (tid < 128) {            // softmax row sums
    float s = 0.f;
    for (int j = 0; j < 128; ++j) s += (float)Kr[tid * 128 + j];
    rowsum[tid] = s;
  }
  __syncthreads();
  for (int idx = tid; idx < 128 * 128; idx += 256)
    Kr[idx] = (_Float16)((float)Kr[idx] / rowsum[idx >> 7]);
  __syncthreads();
  if (tid < 128) {            // column sums (||K||_1); ||K||_inf == 1 (row-stochastic)
    float s = 0.f;
    for (int r = 0; r < 128; ++r) s += (float)Kr[r * 128 + tid];
    colsum[tid] = s;
  }
  __syncthreads();
  if (tid == 0) {
    float mx = colsum[0];
    for (int j = 1; j < 128; ++j) mx = fmaxf(mx, colsum[j]);
    sscal = 1.f / mx;
  }
  __syncthreads();
  for (int idx = tid; idx < 128 * 128; idx += 256) {   // Z0 = K^T / ||K||_1
    int i = idx >> 7, j = idx & 127;
    BufA[idx] = (_Float16)((float)Kr[j * 128 + i] * sscal);
  }
  __syncthreads();
  _Float16* Z = BufA; _Float16* Znx = BufC;
  for (int it = 0; it < NS_ITERS; ++it) {
    mm128<0>(BufB, Kr, Z, wave, lane);     // P  = K @ Z
    __syncthreads();
    mm128<1>(Znx, Z, BufB, wave, lane);    // Z' = Z @ (2I - P)
    __syncthreads();
    _Float16* t = Z; Z = Znx; Znx = t;
  }
  // G = Z @ BV  -> store transposed Gt[n][m]
  const _Float16* BVh = BVg + bh * MM * DD;
  _Float16* Gt = Gtg + bh * DD * MM;
  v16h az[4];
#pragma unroll
  for (int kk = 0; kk < 4; ++kk) az[kk] = loadA16(Z, 128, m0, kk * 32, lane);
  for (int tn = 0; tn < 4; ++tn) {
    v8f c = {};
#pragma unroll
    for (int kk = 0; kk < 4; ++kk)
      c = WMMA_F32_F16(az[kk], loadBn16<0>(BVh, DD, kk * 32, tn * 16, lane), c);
    int n = tn * 16 + (lane & 15);
#pragma unroll
    for (int e = 0; e < 8; ++e) Gt[n * 128 + (m0 + e + mo)] = (_Float16)c[e];
  }
}

// out = softmax(q @ k_land^T) @ G per 128-row block; F never hits HBM.
__global__ void k_fout(const float* __restrict__ q, const _Float16* __restrict__ klg,
                       const _Float16* __restrict__ Gtg, float* __restrict__ out) {
  __shared__ _Float16 buf[128 * 128];   // phase 1: q(128x64)+kl(128x64); phase 2: P(128x128)
  __shared__ _Float16 GT [64 * 128];
  __shared__ float rowsum[128];
  int tid = threadIdx.x, lane = tid & 31, wave = tid >> 5;
  int bh = blockIdx.x >> 6, blk = blockIdx.x & 63;
  int r0 = blk * 128;
  const float*    qp  = q   + ((size_t)bh * SEQ + r0) * DD;
  const _Float16* kl  = klg + bh * MM * DD;
  const _Float16* Gts = Gtg + bh * DD * MM;
  for (int idx = tid; idx < 8192; idx += 256) {
    buf[idx]        = (_Float16)(qp[idx] * 0.125f);
    buf[8192 + idx] = kl[idx];
    GT[idx]         = Gts[idx];
  }
  __syncthreads();
  int m0 = wave * 16, mo = (lane & 16) ? 8 : 0;
  v8f cf[8];
  {
    v16h a0 = loadA16(buf, DD, m0, 0, lane);
    v16h a1 = loadA16(buf, DD, m0, 32, lane);
    for (int tn = 0; tn < 8; ++tn) {
      v8f c = {};
      c = WMMA_F32_F16(a0, loadBt16(buf + 8192, DD, tn * 16, 0, lane), c);
      c = WMMA_F32_F16(a1, loadBt16(buf + 8192, DD, tn * 16, 32, lane), c);
      cf[tn] = c;
    }
  }
  __syncthreads();                       // all reads of q/kl done; reuse buf as P
  for (int tn = 0; tn < 8; ++tn) {
    int n = tn * 16 + (lane & 15);
#pragma unroll
    for (int e = 0; e < 8; ++e) buf[(m0 + e + mo) * 128 + n] = (_Float16)__expf(cf[tn][e]);
  }
  __syncthreads();
  if (tid < 128) {
    float s = 0.f;
    for (int j = 0; j < 128; ++j) s += (float)buf[tid * 128 + j];
    rowsum[tid] = s;
  }
  __syncthreads();
  float* outp = out + ((size_t)bh * SEQ + r0) * DD;
  v16h ap[4];
#pragma unroll
  for (int kk = 0; kk < 4; ++kk) ap[kk] = loadA16(buf, 128, m0, kk * 32, lane);
  for (int tn = 0; tn < 4; ++tn) {
    v8f c = {};
#pragma unroll
    for (int kk = 0; kk < 4; ++kk)
      c = WMMA_F32_F16(ap[kk], loadBt16(GT, 128, tn * 16, kk * 32, lane), c);
    int n = tn * 16 + (lane & 15);
#pragma unroll
    for (int e = 0; e < 8; ++e) {
      int m = m0 + e + mo;
      outp[m * DD + n] = c[e] / rowsum[m];
    }
  }
}

// ---- launch ----------------------------------------------------------------

extern "C" void kernel_launch(void* const* d_in, const int* in_sizes, int n_in,
                              void* d_out, int out_size, void* d_ws, size_t ws_size,
                              hipStream_t stream) {
  (void)in_sizes; (void)n_in; (void)out_size; (void)ws_size;
  const float* q = (const float*)d_in[0];
  const float* k = (const float*)d_in[1];
  const float* v = (const float*)d_in[2];
  float* out = (float*)d_out;
  char* ws = (char*)d_ws;
  _Float16* qlg  = (_Float16*)(ws);                       // 1 MB  q landmarks f16
  _Float16* klg  = (_Float16*)(ws + (1u << 20));          // 1 MB  k landmarks f16
  _Float16* Gtg  = (_Float16*)(ws + (2u << 20));          // 1 MB  G transposed f16
  _Float16* BVg  = (_Float16*)(ws + (3u << 20));          // 1 MB  BV f16
  float*    BVacc= (float*)   (ws + (4u << 20));          // 2 MB  BV f32 partials
  float*    rsg  = (float*)   (ws + (6u << 20));          // 32 KB row sums

  int nz = BH * MM * DD + BH * MM;                        // BVacc + rsg (contiguous)
  k_zero     <<<(nz + 255) / 256, 256, 0, stream>>>(BVacc, nz);
  k_landmarks<<<BH * 32,          256, 0, stream>>>(q, k, qlg, klg);
  k_bv       <<<BH * NSPLIT,      256, 0, stream>>>(k, v, qlg, BVacc, rsg);
  k_bvfin    <<<(BH*MM*DD)/256,   256, 0, stream>>>(BVacc, rsg, BVg);
  k_amat     <<<BH,               256, 0, stream>>>(qlg, klg, BVg, Gtg);
  k_fout     <<<BH * 64,          256, 0, stream>>>(q, klg, Gtg, out);
}
